// OutlookAttention_39539468927741
// MI455X (gfx1250) — matrix operand order repaired and checked by reading, not compiled, not run on hardware
//
#include <hip/hip_runtime.h>

// ---------------- problem constants ----------------
#define Cc    384
#define NHEAD 12
#define HD    32
#define Ee    972        // K^4 * HEADS
#define EPAD  992        // Ee padded to multiple of 32 for WMMA M-tiles
#define Bb    8
#define HH    56
#define LL    3136       // 56*56
#define NT64  49         // LL / 64

typedef __attribute__((ext_vector_type(8)))  _Float16 v8h;
typedef __attribute__((ext_vector_type(16))) _Float16 v16h;
typedef __attribute__((ext_vector_type(8)))  float    v8f;

union AFrag { v16h v; v8h h[2]; };
union HPack { v8h v; _Float16 e[8]; };

// ---------------- f32 -> f16 weight convert (with zero pad) ----------------
__global__ void cvt_f32_f16(const float* __restrict__ s, _Float16* __restrict__ d,
                            int n, int npad) {
    int i = blockIdx.x * 256 + threadIdx.x;
    if (i < npad) d[i] = (i < n) ? (_Float16)s[i] : (_Float16)0.0f;
}

// x[B][C][L] (f32) -> Xt[B][L][C] (f16), coalesced reads over l
__global__ void pack_x(const float* __restrict__ x, _Float16* __restrict__ Xt) {
    int i = blockIdx.x * 256 + threadIdx.x;        // total Bb*Cc*LL (exact)
    int l = i % LL;
    int r = i / LL;
    int c = r % Cc;
    int b = r / Cc;
    Xt[((size_t)b * LL + l) * Cc + c] = (_Float16)x[i];
}

// ---------------- WMMA GEMM: D[M x L] = W[M x 384] @ Xt^T, per batch ----------
// One wave owns a 32(M) x 64(N) tile: 2 A-frags x 4 B-frags -> 8 accumulators.
// MODE 0: write f16 transposed  out16[b][l][m]           (value proj -> Vt)
// MODE 1: write f32 + bias      out32[b][m][l]           (final projection)
// MODE 2: write f16 + bias      out16[b][m][l]           (attention logits)
template <int MODE>
__global__ void __launch_bounds__(256)
gemm16(const _Float16* __restrict__ W, const _Float16* __restrict__ X,
       const float* __restrict__ bias, float* __restrict__ out32,
       _Float16* __restrict__ out16, int Mtot, int nMt) {
    const int lane  = threadIdx.x;            // 0..31
    const int wid   = threadIdx.y;            // 0..7
    const int b     = blockIdx.z;
    const int t     = blockIdx.x * 8 + wid;   // 32x64 tile id
    const int nTile = nMt * NT64;
    if (t >= nTile) return;                   // whole-wave uniform guard
    const int m0 = (t / NT64) * 32;
    const int n0 = (t % NT64) * 64;
    const int laneM = lane & 15;
    const int hi    = lane >> 4;

    const _Float16* __restrict__ Xb = X + (size_t)b * LL * Cc;
    v8f acc[2][4] = {};

    for (int kt = 0; kt < Cc; kt += 32) {
        AFrag a[2], bf[4];
#pragma unroll
        for (int mi = 0; mi < 2; ++mi) {
            // 16-bit A layout: lane holds row M=lane&15; K chunks [8*hi..+7], [16+8*hi..+7]
            const _Float16* wr = W + (size_t)(m0 + mi * 16 + laneM) * Cc + kt;
            a[mi].h[0] = *(const v8h*)(wr + 8 * hi);
            a[mi].h[1] = *(const v8h*)(wr + 16 + 8 * hi);
            if (kt + 32 < Cc) __builtin_prefetch(wr + 32, 0, 1);   // global_prefetch_b8
        }
#pragma unroll
        for (int ni = 0; ni < 4; ++ni) {
            // 16-bit B layout: lane holds col N=lane&15; K chunks [8*hi..+7], [16+8*hi..+7]
            const _Float16* xr = Xb + (size_t)(n0 + ni * 16 + laneM) * Cc + kt;
            bf[ni].h[0] = *(const v8h*)(xr + 8 * hi);
            bf[ni].h[1] = *(const v8h*)(xr + 16 + 8 * hi);
            if (kt + 32 < Cc) __builtin_prefetch(xr + 32, 0, 1);
        }
#pragma unroll
        for (int mi = 0; mi < 2; ++mi)
#pragma unroll
            for (int ni = 0; ni < 4; ++ni)
                acc[mi][ni] = __builtin_amdgcn_wmma_f32_16x16x32_f16(
                    false, a[mi].v, false, bf[ni].v,
                    (short)0, acc[mi][ni], false, false);
    }

#pragma unroll
    for (int mi = 0; mi < 2; ++mi) {
#pragma unroll
        for (int ni = 0; ni < 4; ++ni) {
            const int col = n0 + ni * 16 + laneM;   // N = l
            if (MODE == 0) {
                HPack p;
#pragma unroll
                for (int r = 0; r < 8; ++r) p.e[r] = (_Float16)acc[mi][ni][r];
                // D layout: VGPR r, lane -> (M = r + 8*hi, N = laneM): 8 contiguous c
                _Float16* dst = out16 + ((size_t)b * LL + col) * Cc
                                + (m0 + mi * 16 + 8 * hi);
                *(v8h*)dst = p.v;
            } else if (MODE == 1) {
#pragma unroll
                for (int r = 0; r < 8; ++r) {
                    const int row = m0 + mi * 16 + 8 * hi + r;
                    if (row < Mtot)
                        out32[((size_t)b * Mtot + row) * LL + col] =
                            acc[mi][ni][r] + bias[row];
                }
            } else {
#pragma unroll
                for (int r = 0; r < 8; ++r) {
                    const int row = m0 + mi * 16 + 8 * hi + r;
                    if (row < Mtot)
                        out16[((size_t)b * Mtot + row) * LL + col] =
                            (_Float16)(acc[mi][ni][r] + bias[row]);
                }
            }
        }
    }
}

// -------- softmax over q (groups of 9, stride LL), f16 in/out, f32 math -----
__global__ void softmax9(_Float16* __restrict__ A) {
    int i = blockIdx.x * 256 + threadIdx.x;   // total Bb*NHEAD*9*LL (exact)
    int l = i % LL;
    int r = i / LL;
    int p = r % 9;  r /= 9;
    int n = r % NHEAD;
    int b = r / NHEAD;
    _Float16* base = A + ((size_t)b * Ee + n * 81 + p * 9) * LL + l;
    const float sc = 0.17677669529663687f;    // 32^-0.5
    float v[9], mx = -1e30f;
#pragma unroll
    for (int q = 0; q < 9; ++q) {
        v[q] = (float)base[(size_t)q * LL] * sc;
        mx = fmaxf(mx, v[q]);
    }
    float s = 0.f;
#pragma unroll
    for (int q = 0; q < 9; ++q) { v[q] = __expf(v[q] - mx); s += v[q]; }
    const float inv = 1.f / s;
#pragma unroll
    for (int q = 0; q < 9; ++q) base[(size_t)q * LL] = (_Float16)(v[q] * inv);
}

// ------- fused attention-apply + fold: Ft[b][y*56+x][c] = folded value -------
// folded[b,n*32+d,y,x] = sum_{i,j valid} sum_{q valid}
//     A[b, n*81+(i3+j)*9+q, (y+1-i)*56+(x+1-j)] * V[b, n*32+d, (x+qj-j)*56+(y+qi-i)]
// (the reference's bdwh spatial-transpose quirk is baked into the V column index)
__global__ void __launch_bounds__(256)
apply_fold(const _Float16* __restrict__ A, const _Float16* __restrict__ Vt,
           _Float16* __restrict__ Ft) {
    const int lane = threadIdx.x;                        // d in [0,32)
    const int w    = blockIdx.x * 8 + threadIdx.y;       // wave id (exact grid)
    const int pos  = w % LL;
    int r          = w / LL;
    const int n    = r % NHEAD;
    const int b    = r / NHEAD;
    const int y = pos / HH, x = pos % HH;

    const _Float16* __restrict__ Vb = Vt + (size_t)b * LL * Cc + n * HD + lane;
    const _Float16* __restrict__ Ab = A + ((size_t)b * Ee + n * 81) * LL;

    float acc = 0.f;
#pragma unroll
    for (int i = 0; i < 3; ++i) {
#pragma unroll
        for (int j = 0; j < 3; ++j) {
            const int hp = y + 1 - i, wp = x + 1 - j;    // source l' for this fold shift
            if (hp < 0 || hp >= HH || wp < 0 || wp >= HH) continue;
            const _Float16* arow = Ab + (size_t)((i * 3 + j) * 9) * LL + hp * HH + wp;
#pragma unroll
            for (int qi = 0; qi < 3; ++qi) {
                const int vy = y + qi - i;
                if (vy < 0 || vy >= HH) continue;
#pragma unroll
                for (int qj = 0; qj < 3; ++qj) {
                    const int vx = x + qj - j;
                    if (vx < 0 || vx >= HH) continue;
                    const float av = (float)arow[(size_t)(qi * 3 + qj) * LL];
                    const float vv = (float)Vb[(size_t)(vx * HH + vy) * Cc];
                    acc += av * vv;
                }
            }
        }
    }
    Ft[((size_t)b * LL + pos) * Cc + n * HD + lane] = (_Float16)acc;
}

// ---------------------------------------------------------------------------
extern "C" void kernel_launch(void* const* d_in, const int* in_sizes, int n_in,
                              void* d_out, int out_size, void* d_ws, size_t ws_size,
                              hipStream_t stream) {
    const float* x  = (const float*)d_in[0];
    const float* Wv = (const float*)d_in[1];
    const float* Wa = (const float*)d_in[2];
    const float* ba = (const float*)d_in[3];
    const float* Wp = (const float*)d_in[4];
    const float* bp = (const float*)d_in[5];
    float* out = (float*)d_out;

    char* ws = (char*)d_ws;
    size_t off = 0;
    auto carve = [&](size_t bytes) -> void* {
        void* p = ws + off;
        off += (bytes + 255) & ~(size_t)255;
        return p;
    };
    _Float16* Xt   = (_Float16*)carve((size_t)Bb * LL * Cc * 2);  // 19.3 MB
    _Float16* Wvh  = (_Float16*)carve((size_t)Cc * Cc * 2);
    _Float16* Wah  = (_Float16*)carve((size_t)EPAD * Cc * 2);
    _Float16* Wph  = (_Float16*)carve((size_t)Cc * Cc * 2);
    _Float16* Vt   = (_Float16*)carve((size_t)Bb * LL * Cc * 2);  // 19.3 MB
    _Float16* Abuf = (_Float16*)carve((size_t)Bb * Ee * LL * 2);  // 48.8 MB
    _Float16* Ft   = (_Float16*)carve((size_t)Bb * LL * Cc * 2);  // 19.3 MB

    // 1) convert weights (Wa zero-padded to 992 rows)
    cvt_f32_f16<<<(Cc * Cc + 255) / 256, 256, 0, stream>>>(Wv, Wvh, Cc * Cc, Cc * Cc);
    cvt_f32_f16<<<(EPAD * Cc + 255) / 256, 256, 0, stream>>>(Wa, Wah, Ee * Cc, EPAD * Cc);
    cvt_f32_f16<<<(Cc * Cc + 255) / 256, 256, 0, stream>>>(Wp, Wph, Cc * Cc, Cc * Cc);

    // 2) pack x into [b][l][c] f16
    pack_x<<<(Bb * Cc * LL) / 256, 256, 0, stream>>>(x, Xt);

    dim3 gblk(32, 8);
    // 3) value projection -> Vt[b][l][c]  (12 M-tiles x 49 N-tiles = 588 -> 74 blocks)
    gemm16<0><<<dim3(74, 1, Bb), gblk, 0, stream>>>(Wvh, Xt, nullptr, nullptr, Vt, Cc, 12);
    // 4) attention logits + bias (f16) -> Abuf[b][e][l]  (31 x 49 = 1519 -> 190 blocks)
    gemm16<2><<<dim3(190, 1, Bb), gblk, 0, stream>>>(Wah, Xt, ba, nullptr, Abuf, Ee, 31);
    // 5) softmax over q, in place (f16 storage, f32 math)
    softmax9<<<(Bb * NHEAD * 9 * LL) / 256, 256, 0, stream>>>(Abuf);
    // 6) fused apply + fold -> Ft[b][l][c]
    apply_fold<<<dim3((Bb * NHEAD * LL) / 8), gblk, 0, stream>>>(Abuf, Vt, Ft);
    // 7) final projection + bias -> out[b][d][l]
    gemm16<1><<<dim3(74, 1, Bb), gblk, 0, stream>>>(Wph, Ft, bp, out, nullptr, Cc, 12);
}
